// TrustCompletionModel_2138893714144
// MI455X (gfx1250) — matrix operand order repaired and checked by reading, not compiled, not run on hardware
//
#include <hip/hip_runtime.h>
#include <hip/hip_bf16.h>
#include <stdint.h>

typedef __attribute__((ext_vector_type(16))) _Float16 v16h;
typedef __attribute__((ext_vector_type(8)))  float    v8f;

#define MDIM 4096
#define GAT_NEG_INF (-9e15f)
#define GAT_ALPHA 0.2f

// ---------------------------------------------------------------------------
// Pack adj (int32 0/1, row-major [M][M]) into bitmask words [M][M/32].
// wave32: __ballot bit i == lane i, lanes map to consecutive columns.
// ---------------------------------------------------------------------------
__global__ void pack_adj_kernel(const int* __restrict__ adj,
                                uint32_t* __restrict__ adjp) {
    int idx = blockIdx.x * blockDim.x + threadIdx.x;
    int pred = adj[idx] != 0;
    uint32_t mask = (uint32_t)__ballot(pred);
    if ((threadIdx.x & 31) == 0) adjp[idx >> 5] = mask;
}

// ---------------------------------------------------------------------------
// Wh = hin[M x K] @ W[K x F]   (f32, tiny GEMM — exact precision on weights)
// Also emits WhT in f16, [F x M], so the attention kernel's B-operand gather
// is contiguous (one 32B load per lane per N-tile half).
// ---------------------------------------------------------------------------
__global__ void gemm_wh_kernel(const float* __restrict__ hin,
                               const float* __restrict__ W,
                               float* __restrict__ Wh,
                               _Float16* __restrict__ WhT,
                               int K, int F) {
    int idx = blockIdx.x * blockDim.x + threadIdx.x;
    int i = idx / F, j = idx % F;
    float acc = 0.f;
    for (int k = 0; k < K; ++k)
        acc = fmaf(hin[i * K + k], W[k * F + j], acc);
    Wh[i * F + j] = acc;
    WhT[(size_t)j * MDIM + i] = (_Float16)acc;
}

// ---------------------------------------------------------------------------
// s_i = Wh @ a[0:F], s_k = Wh @ a[F:2F]   (one thread per row)
// ---------------------------------------------------------------------------
__global__ void score_kernel(const float* __restrict__ Wh,
                             const float* __restrict__ a,
                             float* __restrict__ si,
                             float* __restrict__ sk, int F) {
    int i = blockIdx.x * blockDim.x + threadIdx.x;
    float accI = 0.f, accK = 0.f;
    for (int j = 0; j < F; ++j) {
        float w = Wh[i * F + j];
        accI = fmaf(w, a[j], accI);
        accK = fmaf(w, a[F + j], accK);
    }
    si[i] = accI;
    sk[i] = accK;
}

// ---------------------------------------------------------------------------
// Fused masked-softmax(row) @ Wh with online (flash) softmax and WMMA f16.
// One wave per 16-row tile. Per 32-wide k-chunk:
//   - each lane computes its 16 e-values directly in A-operand order (contiguous k, matching B):
//       lane L   : row L, cols {0..15}
//       lane L+16: row L, cols {16..31}
//   - online max/sum via __shfl_xor(16); probs -> f16 A operand in-register
//   - B operand: 16 contiguous f16 from WhT (column segment), per N-tile
//   - accumulator rescale (shuffle-broadcast of per-row factor into the C/D
//     layout, VGPR r <-> row r / r+8) is guarded by a wave-uniform ballot:
//     the running max rarely changes, so the rescale is usually skipped.
// Epilogue: one reciprocal per lane, broadcast, multiply; elu; store f32.
// ---------------------------------------------------------------------------
template <int F>
__global__ __launch_bounds__(128) void gat_attn_kernel(
    const uint32_t* __restrict__ adjp,
    const float* __restrict__ si,
    const float* __restrict__ sk,
    const _Float16* __restrict__ WhT,
    float* __restrict__ hout) {
    constexpr int NG = F / 16;
    const int lane = threadIdx.x & 31;
    const int wave = threadIdx.x >> 5;
    const int itile = blockIdx.x * 4 + wave;
    const int rloc = lane & 15;
    const bool hi = lane >= 16;
    const int row = itile * 16 + rloc;

    const float srow = si[row];
    const uint32_t* adjrow = adjp + (size_t)row * (MDIM / 32);

    float mx = -__builtin_inff();
    float sm = 0.f;
    v8f acc[NG] = {};

    for (int k0 = 0; k0 < MDIM; k0 += 32) {
        uint32_t bits = adjrow[k0 >> 5];

        float e[16];
        float tmax = -__builtin_inff();
#pragma unroll
        for (int j = 0; j < 16; ++j) {
            int c = j + (hi ? 16 : 0);
            float t = srow + sk[k0 + c];
            t = t > 0.f ? t : GAT_ALPHA * t;            // leaky relu
            float ev = ((bits >> c) & 1u) ? t : GAT_NEG_INF;
            e[j] = ev;
            tmax = fmaxf(tmax, ev);
        }
        tmax = fmaxf(tmax, __shfl_xor(tmax, 16, 32));    // combine row halves
        const float nmx = fmaxf(mx, tmax);
        const bool upd = nmx > mx;                       // row max moved?
        const float f = upd ? __expf(mx - nmx) : 1.f;    // first iter: exp(-inf)=0
        mx = nmx;

        float tsum = 0.f;
        v16h A;
#pragma unroll
        for (int j = 0; j < 16; ++j) {
            float p = __expf(e[j] - nmx);
            tsum += p;
            A[j] = (_Float16)p;                          // A-layout order by design
        }
        tsum += __shfl_xor(tsum, 16, 32);
        sm = sm * f + tsum;                              // f==1 when max unchanged

        // Rescale only if any row's max actually moved (wave-uniform branch;
        // expected O(log #tiles) times over the whole loop).
        if (__ballot(upd) != 0ull) {
            float fr[8];
#pragma unroll
            for (int r = 0; r < 8; ++r) fr[r] = __shfl(f, (hi ? 8 : 0) + r, 32);
#pragma unroll
            for (int g = 0; g < NG; ++g) {
#pragma unroll
                for (int r = 0; r < 8; ++r) acc[g][r] *= fr[r];
            }
        }

        const int kk = k0 + (hi ? 16 : 0);
#pragma unroll
        for (int g = 0; g < NG; ++g) {
            int n = g * 16 + rloc;
            v16h B = *(const v16h*)(WhT + (size_t)n * MDIM + kk);
            acc[g] = __builtin_amdgcn_wmma_f32_16x16x32_f16(
                false, A, false, B, (short)0, acc[g], false, false);
        }
    }

    // epilogue: one reciprocal per lane, broadcast into C/D layout, elu, store
    const float rs = 1.0f / sm;
    float smr[8];
#pragma unroll
    for (int r = 0; r < 8; ++r) smr[r] = __shfl(rs, (hi ? 8 : 0) + r, 32);
#pragma unroll
    for (int g = 0; g < NG; ++g) {
#pragma unroll
        for (int r = 0; r < 8; ++r) {
            float v = acc[g][r] * smr[r];
            v = v > 0.f ? v : (__expf(v) - 1.f);         // elu
            hout[(size_t)(itile * 16 + r + (hi ? 8 : 0)) * F + g * 16 + rloc] = v;
        }
    }
}

// ---------------------------------------------------------------------------
// pred[i,k] = sigmoid(p_i[i] + p_k[k]), float4 stores (write-bandwidth bound)
// ---------------------------------------------------------------------------
__global__ void pred_kernel(const float* __restrict__ pi,
                            const float* __restrict__ pk,
                            float* __restrict__ out) {
    int idx = blockIdx.x * blockDim.x + threadIdx.x;    // one float4 each
    int base = idx * 4;
    int i = base / MDIM;
    int k = base % MDIM;
    float a = pi[i];
    float4 v;
    v.x = 1.f / (1.f + __expf(-(a + pk[k + 0])));
    v.y = 1.f / (1.f + __expf(-(a + pk[k + 1])));
    v.z = 1.f / (1.f + __expf(-(a + pk[k + 2])));
    v.w = 1.f / (1.f + __expf(-(a + pk[k + 3])));
    ((float4*)out)[idx] = v;
}

// ---------------------------------------------------------------------------
// Host-side launch
// inputs: h0[M,64] f32, adj_mask[M,M] i32, W1[64,64], a1[128,1], W2[64,32],
//         a2[64,1], omega[64,1]
// output: pred[M,M] f32 ++ hL[M,32] f32
// ---------------------------------------------------------------------------
extern "C" void kernel_launch(void* const* d_in, const int* in_sizes, int n_in,
                              void* d_out, int out_size, void* d_ws, size_t ws_size,
                              hipStream_t stream) {
    const float* h0    = (const float*)d_in[0];
    const int*   adj   = (const int*)d_in[1];
    const float* W1    = (const float*)d_in[2];
    const float* a1    = (const float*)d_in[3];
    const float* W2    = (const float*)d_in[4];
    const float* a2    = (const float*)d_in[5];
    const float* omega = (const float*)d_in[6];
    float* out = (float*)d_out;

    const size_t M = MDIM;
    uint8_t* w = (uint8_t*)d_ws;
    size_t off = 0;
    auto alloc = [&](size_t bytes) {
        void* p = w + off;
        off = (off + bytes + 255) & ~(size_t)255;
        return p;
    };
    uint32_t* adjp  = (uint32_t*)alloc(M * M / 8);        // 2 MB bitmask
    float*    Wh1   = (float*)alloc(M * 64 * 4);
    _Float16* WhT1  = (_Float16*)alloc(64 * M * 2);
    float*    s1i   = (float*)alloc(M * 4);
    float*    s1k   = (float*)alloc(M * 4);
    float*    h1    = (float*)alloc(M * 64 * 4);
    float*    Wh2   = (float*)alloc(M * 32 * 4);
    _Float16* WhT2  = (_Float16*)alloc(32 * M * 2);
    float*    s2i   = (float*)alloc(M * 4);
    float*    s2k   = (float*)alloc(M * 4);
    float*    p_i   = (float*)alloc(M * 4);
    float*    p_k   = (float*)alloc(M * 4);
    (void)ws_size; (void)in_sizes; (void)n_in; (void)out_size;

    float* hL = out + M * M;   // layer-2 attention writes hL directly to output tail

    // adjacency bit-pack (67 MB -> 2 MB; read once)
    pack_adj_kernel<<<M * M / 256, 256, 0, stream>>>(adj, adjp);

    // ----- layer 1 -----
    gemm_wh_kernel<<<M * 64 / 256, 256, 0, stream>>>(h0, W1, Wh1, WhT1, 64, 64);
    score_kernel<<<M / 256, 256, 0, stream>>>(Wh1, a1, s1i, s1k, 64);
    gat_attn_kernel<64><<<M / 64, 128, 0, stream>>>(adjp, s1i, s1k, WhT1, h1);

    // ----- layer 2 -----
    gemm_wh_kernel<<<M * 32 / 256, 256, 0, stream>>>(h1, W2, Wh2, WhT2, 64, 32);
    score_kernel<<<M / 256, 256, 0, stream>>>(Wh2, a2, s2i, s2k, 32);
    gat_attn_kernel<32><<<M / 64, 128, 0, stream>>>(adjp, s2i, s2k, WhT2, hL);

    // ----- link prediction -----
    score_kernel<<<M / 256, 256, 0, stream>>>(hL, omega, p_i, p_k, 32);
    pred_kernel<<<M * M / 4 / 256, 256, 0, stream>>>(p_i, p_k, out);
}
